// PointNetV2SemanticSegmentation_30288109371627
// MI455X (gfx1250) — compile-verified, matrix operand order checked by reading, not run on hardware
//
#include <hip/hip_runtime.h>
#include <hip/hip_bf16.h>

// PointNet++ semantic segmentation forward, CDNA5 (gfx1250, wave32).
// Dense MLP layers -> v_wmma_f32_16x16x32_f16 tiled GEMM. All GEMM dims are
// statically padded (weights pre-padded into workspace; M always a multiple
// of 128 for this network), so the hot loop is branch-free with pointer
// increments only: wide global loads -> packed DS stores -> ds_load_b128
// fragments -> WMMA.

typedef __attribute__((ext_vector_type(16))) _Float16 v16h;
typedef __attribute__((ext_vector_type(8)))  _Float16 v8h;
typedef __attribute__((ext_vector_type(4)))  _Float16 v4h;
typedef __attribute__((ext_vector_type(8)))  float    v8f;

#define BATCH 8
#define NPTS  4096
#define KSAMP 32

// ---------------------------------------------------------------------------
// WMMA GEMM: Y[M,N] = act(X[M,K] @ W[K,N] + bias[N]).
// Preconditions (guaranteed by host): M % 128 == 0, K % 32 == 0 (X stride K,
// zero-padded cols), N % 16 == 0 (W/bias/Y padded). No masking anywhere.
// Block: 256 thr = 8 waves; wave w owns C tile [(bx*8+w)*16, by*16].
// ---------------------------------------------------------------------------
__global__ __launch_bounds__(256)
void k_gemm_wmma(const float* __restrict__ X, const float* __restrict__ W,
                 const float* __restrict__ bias, float* __restrict__ Y,
                 int M, int K, int N, int relu)
{
    __shared__ __align__(16) _Float16 sA[8][16 * 32];   // [m][k] per wave
    __shared__ __align__(16) _Float16 sWt[16 * 32];     // [n][k] transposed

    const int lane = threadIdx.x & 31;
    const int wave = threadIdx.x >> 5;
    const int m0   = (blockIdx.x * 8 + wave) * 16;
    const int n0   = blockIdx.y * 16;
    const int half = lane >> 4;             // 0: lanes 0-15, 1: lanes 16-31
    const int l16  = lane & 15;
    const int tw   = threadIdx.x;

    union { v8f v; float f[8]; } acc;
#pragma unroll
    for (int i = 0; i < 8; ++i) acc.f[i] = 0.0f;

    // Loop-carried pointers (no per-iteration multiplies).
    // W staging: thread tw owns k-locals (tw>>4) and (tw>>4)+16, col tw&15.
    const float* wp = W + (size_t)(tw >> 4) * N + n0 + (tw & 15);
    _Float16*   sw0 = &sWt[(tw & 15) * 32 + (tw >> 4)];
    // A staging: lane owns float4-column (lane&7), rows (lane>>3)+{0,4,8,12}.
    const float* ap = X + (size_t)(m0 + (lane >> 3)) * K + (lane & 7) * 4;
    _Float16*   sa0 = &sA[wave][(lane >> 3) * 32 + (lane & 7) * 4];
    const size_t wstep16 = (size_t)16 * N;      // 16 W-rows (floats)
    const size_t astep4  = (size_t)4 * K;       // 4 X-rows (floats)

    const int nsteps = K >> 5;
    for (int s = 0; s < nsteps; ++s) {
        // W chunk -> LDS transposed [n][k]
        sw0[0]  = (_Float16)wp[0];
        sw0[16] = (_Float16)wp[wstep16];
        // A tile -> LDS [m][k]: 4x float4 load + packed ds_store_b64
#pragma unroll
        for (int j = 0; j < 4; ++j) {
            const float4 vx = *(const float4*)(ap + (size_t)j * astep4);
            v4h h;
            h.x = (_Float16)vx.x; h.y = (_Float16)vx.y;
            h.z = (_Float16)vx.z; h.w = (_Float16)vx.w;
            *(v4h*)(sa0 + j * 128) = h;
        }
        __builtin_prefetch(ap + 32, 0, 1);      // speculative next chunk
        __syncthreads();

        // Fragments: f16 16x32 layout -> two contiguous 8-half runs per lane.
        union { v16h v; v8h q[2]; } a, b;
        a.q[0] = *(const v8h*)&sA[wave][l16 * 32 + half * 8];
        a.q[1] = *(const v8h*)&sA[wave][l16 * 32 + 16 + half * 8];
        b.q[0] = *(const v8h*)&sWt[l16 * 32 + half * 8];
        b.q[1] = *(const v8h*)&sWt[l16 * 32 + 16 + half * 8];
        acc.v = __builtin_amdgcn_wmma_f32_16x16x32_f16(
            false, a.v, false, b.v, (short)0, acc.v, false, false);
        __syncthreads();

        wp += (size_t)32 * N;
        ap += 32;
    }

    // C/D layout: lane L, VGPR v -> M = v + (L/16)*8, N = L%16. Unmasked.
    const float bb = bias[n0 + l16];
    float* yp = Y + (size_t)(m0 + (half << 3)) * N + n0 + l16;
#pragma unroll
    for (int v = 0; v < 8; ++v) {
        float y = acc.f[v] + bb;
        if (relu) y = y > 0.0f ? y : 0.0f;
        yp[(size_t)v * N] = y;
    }
}

// Pad weights W[Kw,N] -> Wp[Kp,Np] (zero fill), bias[N] -> bp[Np].
__global__ void k_pad_w(const float* __restrict__ W, const float* __restrict__ b,
                        int Kw, int N, int Kp, int Np,
                        float* __restrict__ Wp, float* __restrict__ bp)
{
    int t = blockIdx.x * blockDim.x + threadIdx.x;
    int tot = Kp * Np;
    if (t < tot) {
        int k = t / Np, n = t % Np;
        Wp[t] = (k < Kw && n < N) ? W[(size_t)k * N + n] : 0.0f;
    }
    if (t < Np) bp[t] = (t < N) ? b[t] : 0.0f;
}

// x[B,3,N] -> xyz[B,N,3]
__global__ void k_transpose_in(const float* __restrict__ x,
                               float* __restrict__ xyz, int N, int total)
{
    int t = blockIdx.x * blockDim.x + threadIdx.x;
    if (t >= total) return;
    int c = t % 3, bn = t / 3;
    int n = bn % N, b = bn / N;
    xyz[t] = x[((size_t)b * 3 + c) * N + n];
}

// Farthest point sampling: one block per batch, dist in LDS, block argmax.
__global__ __launch_bounds__(256)
void k_fps(const float* __restrict__ xyz, int N, int npoint, int* __restrict__ out)
{
    __shared__ float dist[NPTS];
    __shared__ float rv[256];
    __shared__ int   ri[256];
    __shared__ int   s_far;

    const int b = blockIdx.x;
    const float* p = xyz + (size_t)b * N * 3;
    int* o = out + (size_t)b * npoint;

    for (int i = threadIdx.x; i < N; i += 256) dist[i] = 1e10f;
    if (threadIdx.x == 0) s_far = 0;
    __syncthreads();

    for (int j = 0; j < npoint; ++j) {
        int far = s_far;
        if (threadIdx.x == 0) o[j] = far;
        float cx = p[far * 3 + 0], cy = p[far * 3 + 1], cz = p[far * 3 + 2];
        float bv = -1.0f; int bi = 0;
        for (int i = threadIdx.x; i < N; i += 256) {
            float dx = p[i * 3 + 0] - cx;
            float dy = p[i * 3 + 1] - cy;
            float dz = p[i * 3 + 2] - cz;
            float d  = dx * dx + dy * dy + dz * dz;
            float m  = dist[i];
            m = m < d ? m : d;
            dist[i] = m;
            if (m > bv) { bv = m; bi = i; }
        }
        rv[threadIdx.x] = bv; ri[threadIdx.x] = bi;
        __syncthreads();
        for (int s = 128; s > 0; s >>= 1) {
            if (threadIdx.x < s) {
                if (rv[threadIdx.x + s] > rv[threadIdx.x]) {
                    rv[threadIdx.x] = rv[threadIdx.x + s];
                    ri[threadIdx.x] = ri[threadIdx.x + s];
                }
            }
            __syncthreads();
        }
        if (threadIdx.x == 0) s_far = ri[0];
        __syncthreads();
    }
}

// gather 3-vectors: out[b,s,:] = xyz[b, idx[b,s], :]
__global__ void k_gather3(const float* __restrict__ xyz, const int* __restrict__ idx,
                          int N, int S, float* __restrict__ out, int total)
{
    int t = blockIdx.x * blockDim.x + threadIdx.x;
    if (t >= total) return;                 // t = b*S + s
    int b = t / S;
    int id = idx[t];
    const float* p = xyz + ((size_t)b * N + id) * 3;
    out[(size_t)t * 3 + 0] = p[0];
    out[(size_t)t * 3 + 1] = p[1];
    out[(size_t)t * 3 + 2] = p[2];
}

// Ball query: first KSAMP indices (ascending) with sqdist <= r2, pad w/ first.
__global__ void k_query_ball(const float* __restrict__ xyz, const float* __restrict__ cen,
                             int N, int S, float r2, int* __restrict__ gidx, int total)
{
    int t = blockIdx.x * blockDim.x + threadIdx.x;
    if (t >= total) return;                 // t = b*S + s
    int b = t / S;
    const float* p = xyz + (size_t)b * N * 3;
    const float* c = cen + (size_t)t * 3;
    int* o = gidx + (size_t)t * KSAMP;
    int cnt = 0, first = 0; bool has = false;
    for (int n = 0; n < N && cnt < KSAMP; ++n) {
        float dx = p[n * 3 + 0] - c[0];
        float dy = p[n * 3 + 1] - c[1];
        float dz = p[n * 3 + 2] - c[2];
        if (dx * dx + dy * dy + dz * dz <= r2) {
            if (!has) { first = n; has = true; }
            o[cnt++] = n;
        }
    }
    for (; cnt < KSAMP; ++cnt) o[cnt] = first;
}

// Grouped GEMM input rows: [centered xyz | gathered feats | zero pad to Kpad]
__global__ void k_group(const float* __restrict__ xyz, const float* __restrict__ cen,
                        const float* __restrict__ feats, const int* __restrict__ gidx,
                        int N, int S, int Cf, int Kpad, float* __restrict__ out,
                        int total)
{
    int t = blockIdx.x * blockDim.x + threadIdx.x;
    if (t >= total) return;                 // t = (b*S+s)*KSAMP + k
    int bs = t / KSAMP;
    int b  = bs / S;
    int id = gidx[t];
    const float* pt = xyz + ((size_t)b * N + id) * 3;
    const float* c  = cen + (size_t)bs * 3;
    float* o = out + (size_t)t * Kpad;
    o[0] = pt[0] - c[0];
    o[1] = pt[1] - c[1];
    o[2] = pt[2] - c[2];
    if (feats) {
        const float* f = feats + ((size_t)b * N + id) * Cf;
        for (int cc = 0; cc < Cf; ++cc) o[3 + cc] = f[cc];
    }
    for (int cc = 3 + Cf; cc < Kpad; ++cc) o[cc] = 0.0f;
}

// Max-pool over the KSAMP group dimension: in[(r*K+k), c] -> out[r, c]
__global__ void k_maxpool(const float* __restrict__ in, float* __restrict__ out,
                          int C, int total)
{
    int t = blockIdx.x * blockDim.x + threadIdx.x;
    if (t >= total) return;                 // t = r*C + c
    int r = t / C, c = t % C;
    const float* p = in + (size_t)r * KSAMP * C + c;
    float m = p[0];
    for (int k = 1; k < KSAMP; ++k) {
        float v = p[(size_t)k * C];
        m = v > m ? v : m;
    }
    out[t] = m;
}

// 3-NN of xyz1 points among xyz2 points; inverse-distance weights.
__global__ void k_three_nn(const float* __restrict__ xyz1, const float* __restrict__ xyz2,
                           int N1, int N2, int* __restrict__ idx3, float* __restrict__ w3,
                           int total)
{
    int t = blockIdx.x * blockDim.x + threadIdx.x;
    if (t >= total) return;                 // t = b*N1 + n
    int b = t / N1;
    const float* a = xyz1 + (size_t)t * 3;
    const float* q = xyz2 + (size_t)b * N2 * 3;
    float d0 = 1e20f, d1 = 1e20f, d2 = 1e20f;
    int   i0 = 0, i1 = 0, i2 = 0;
    for (int n = 0; n < N2; ++n) {
        float dx = q[n * 3 + 0] - a[0];
        float dy = q[n * 3 + 1] - a[1];
        float dz = q[n * 3 + 2] - a[2];
        float d  = dx * dx + dy * dy + dz * dz;
        if (d < d0)      { d2 = d1; i2 = i1; d1 = d0; i1 = i0; d0 = d; i0 = n; }
        else if (d < d1) { d2 = d1; i2 = i1; d1 = d; i1 = n; }
        else if (d < d2) { d2 = d; i2 = n; }
    }
    float w0 = 1.0f / (fmaxf(d0, 0.0f) + 1e-8f);
    float w1 = 1.0f / (fmaxf(d1, 0.0f) + 1e-8f);
    float w2 = 1.0f / (fmaxf(d2, 0.0f) + 1e-8f);
    float s = w0 + w1 + w2;
    w3[(size_t)t * 3 + 0] = w0 / s;  idx3[(size_t)t * 3 + 0] = i0;
    w3[(size_t)t * 3 + 1] = w1 / s;  idx3[(size_t)t * 3 + 1] = i1;
    w3[(size_t)t * 3 + 2] = w2 / s;  idx3[(size_t)t * 3 + 2] = i2;
}

// out[row, :] = [ f1[row, :D1] | sum_j w_j * f2[idx_j, :D2] ]
__global__ void k_interp_concat(const float* __restrict__ f1, int D1,
                                const float* __restrict__ f2, int D2,
                                const int* __restrict__ idx3, const float* __restrict__ w3,
                                int N2, int N1, float* __restrict__ out, int total)
{
    int t = blockIdx.x * blockDim.x + threadIdx.x;
    if (t >= total) return;                 // t = row*(D1+D2) + c
    int D = D1 + D2;
    int row = t / D, c = t % D;
    if (c < D1) { out[t] = f1[(size_t)row * D1 + c]; return; }
    int cc = c - D1;
    int b  = row / N1;
    const int*   id = idx3 + (size_t)row * 3;
    const float* w  = w3   + (size_t)row * 3;
    const float* F  = f2   + (size_t)b * N2 * D2;
    out[t] = w[0] * F[(size_t)id[0] * D2 + cc]
           + w[1] * F[(size_t)id[1] * D2 + cc]
           + w[2] * F[(size_t)id[2] * D2 + cc];
}

// Row-wise log-softmax over C classes, input stride `ld`.
__global__ void k_log_softmax(const float* __restrict__ in, float* __restrict__ out,
                              int C, int ld, int rows)
{
    int r = blockIdx.x * blockDim.x + threadIdx.x;
    if (r >= rows) return;
    const float* p = in + (size_t)r * ld;
    float m = p[0];
    for (int c = 1; c < C; ++c) m = fmaxf(m, p[c]);
    float s = 0.0f;
    for (int c = 0; c < C; ++c) s += __expf(p[c] - m);
    float ls = __logf(s);
    for (int c = 0; c < C; ++c) out[(size_t)r * C + c] = p[c] - m - ls;
}

// l4f[B,16,512] -> out2[B,512,16]
__global__ void k_out2(const float* __restrict__ l4f, float* __restrict__ out2, int total)
{
    int t = blockIdx.x * blockDim.x + threadIdx.x;
    if (t >= total) return;                 // t = (b*512 + c)*16 + s
    int s  = t % 16;
    int bc = t / 16;
    int c  = bc % 512, b = bc / 512;
    out2[t] = l4f[((size_t)b * 16 + s) * 512 + c];
}

// ---------------------------------------------------------------------------
// Host orchestration
// ---------------------------------------------------------------------------
extern "C" void kernel_launch(void* const* d_in, const int* in_sizes, int n_in,
                              void* d_out, int out_size, void* d_ws, size_t ws_size,
                              hipStream_t stream)
{
    (void)in_sizes; (void)n_in; (void)out_size; (void)ws_size;
    const int Bc = BATCH;
    const float* x = (const float*)d_in[0];

    // ---- bump allocator over workspace ----
    char* wsp = (char*)d_ws;
    size_t off = 0;
    auto alloc_f = [&](size_t nelem) -> float* {
        float* p = (float*)(wsp + off);
        off = (off + nelem * sizeof(float) + 255) & ~(size_t)255;
        return p;
    };
    auto alloc_i = [&](size_t nelem) -> int* {
        int* p = (int*)(wsp + off);
        off = (off + nelem * sizeof(int) + 255) & ~(size_t)255;
        return p;
    };

    float* xyz  = alloc_f((size_t)Bc * NPTS * 3);
    float* l1x  = alloc_f((size_t)Bc * 1024 * 3);
    float* l2x  = alloc_f((size_t)Bc * 256 * 3);
    float* l3x  = alloc_f((size_t)Bc * 64 * 3);
    float* l4x  = alloc_f((size_t)Bc * 16 * 3);
    float* l1f  = alloc_f((size_t)Bc * 1024 * 64);
    float* l2f  = alloc_f((size_t)Bc * 256 * 128);
    float* l3f  = alloc_f((size_t)Bc * 64 * 256);
    float* l4f  = alloc_f((size_t)Bc * 16 * 512);
    float* f3   = alloc_f((size_t)Bc * 64 * 256);
    float* f2_  = alloc_f((size_t)Bc * 256 * 256);
    float* f1_  = alloc_f((size_t)Bc * 1024 * 128);
    float* f0   = alloc_f((size_t)Bc * NPTS * 128);
    int*   fpsI = alloc_i((size_t)Bc * 1024);
    int*   gidx = alloc_i((size_t)Bc * 1024 * KSAMP);
    int*   idx3 = alloc_i((size_t)Bc * NPTS * 3);
    float* w3   = alloc_f((size_t)Bc * NPTS * 3);
    float* inbuf = alloc_f((size_t)262144 * 32);            // max padded GEMM input (sa1)
    float* bufA  = alloc_f((size_t)262144 * 64);            // max GEMM output (sa1)
    float* bufB  = alloc_f((size_t)262144 * 64);

    // ---- pre-pad weights/biases into workspace (Kp % 32, Np % 16) ----
    struct LayerW { int wi, Kw, N; };
    static const LayerW layers[] = {
        {1, 3, 32},   {3, 32, 32},  {5, 32, 64},
        {7, 67, 64},  {9, 64, 64},  {11, 64, 128},
        {13, 131, 128}, {15, 128, 128}, {17, 128, 256},
        {19, 259, 256}, {21, 256, 256}, {23, 256, 512},
        {25, 768, 256}, {27, 256, 256},
        {29, 384, 256}, {31, 256, 256},
        {33, 320, 256}, {35, 256, 128},
        {37, 128, 128}, {39, 128, 128}, {41, 128, 128}, {43, 128, 128},
        {45, 128, 13}
    };
    float* Wp[47] = {nullptr};
    float* Bp[47] = {nullptr};
    int    Kp[47] = {0};
    int    Np[47] = {0};
    for (const LayerW& L : layers) {
        int kp = (L.Kw + 31) & ~31;
        int np = (L.N + 15) & ~15;
        Wp[L.wi] = alloc_f((size_t)kp * np);
        Bp[L.wi] = alloc_f((size_t)np);
        Kp[L.wi] = kp; Np[L.wi] = np;
        int tot = kp * np;
        k_pad_w<<<(tot + 255) / 256, 256, 0, stream>>>(
            (const float*)d_in[L.wi], (const float*)d_in[L.wi + 1],
            L.Kw, L.N, kp, np, Wp[L.wi], Bp[L.wi]);
    }

    auto gemm = [&](const float* X, int M, int wi, float* Y, int relu) {
        int K = Kp[wi], N = Np[wi];
        dim3 g((unsigned)(M / 128), (unsigned)(N / 16));
        k_gemm_wmma<<<g, 256, 0, stream>>>(X, Wp[wi], Bp[wi], Y, M, K, N, relu);
    };

    // MLP chain (all layers ReLU). X0 stride must equal Kp[wis[0]].
    auto mlp = [&](const float* X0, int M, const int* wis, int nl,
                   float* final_out) -> const float* {
        const float* cur = X0;
        for (int i = 0; i < nl; ++i) {
            float* Y = (i == nl - 1 && final_out) ? final_out
                                                  : ((i & 1) ? bufB : bufA);
            gemm(cur, M, wis[i], Y, 1);
            cur = Y;
        }
        return cur;
    };

    auto sa_stage = [&](const float* xin, const float* fin, int Nin, int S,
                        float radius, int Cf, const int* wis, int nl, int Cout,
                        float* xout, float* fout) {
        k_fps<<<Bc, 256, 0, stream>>>(xin, Nin, S, fpsI);
        int tS = Bc * S;
        k_gather3<<<(tS + 255) / 256, 256, 0, stream>>>(xin, fpsI, Nin, S, xout, tS);
        k_query_ball<<<(tS + 255) / 256, 256, 0, stream>>>(xin, xout, Nin, S,
                                                           radius * radius, gidx, tS);
        int M = tS * KSAMP;
        int Kpad = Kp[wis[0]];
        k_group<<<(M + 255) / 256, 256, 0, stream>>>(xin, xout, fin, gidx, Nin, S,
                                                     Cf, Kpad, inbuf, M);
        const float* last = mlp(inbuf, M, wis, nl, nullptr);
        int tp = tS * Cout;
        k_maxpool<<<(tp + 255) / 256, 256, 0, stream>>>(last, fout, Cout, tp);
    };

    auto fp_stage = [&](const float* x1, int N1, const float* x2, int N2,
                        const float* f1p, int D1, const float* f2p, int D2,
                        const int* wis, int nl, float* fout) {
        int t1 = Bc * N1;
        k_three_nn<<<(t1 + 255) / 256, 256, 0, stream>>>(x1, x2, N1, N2, idx3, w3, t1);
        int D = D1 + D2;                    // multiple of 32 for every FP stage
        int tt = t1 * D;
        k_interp_concat<<<(tt + 255) / 256, 256, 0, stream>>>(f1p, D1, f2p, D2,
                                                              idx3, w3, N2, N1,
                                                              inbuf, tt);
        mlp(inbuf, t1, wis, nl, fout);
    };

    // ---- forward pass ----
    {
        int tt = Bc * NPTS * 3;
        k_transpose_in<<<(tt + 255) / 256, 256, 0, stream>>>(x, xyz, NPTS, tt);
    }

    const int sa1_wi[] = {1, 3, 5};
    const int sa2_wi[] = {7, 9, 11};
    const int sa3_wi[] = {13, 15, 17};
    const int sa4_wi[] = {19, 21, 23};
    const int fp4_wi[] = {25, 27};
    const int fp3_wi[] = {29, 31};
    const int fp2_wi[] = {33, 35};
    const int fp1_wi[] = {37, 39, 41, 43};

    sa_stage(xyz, nullptr, NPTS, 1024, 0.1f, 0,   sa1_wi, 3, 64,  l1x, l1f);
    sa_stage(l1x, l1f,     1024, 256,  0.2f, 64,  sa2_wi, 3, 128, l2x, l2f);
    sa_stage(l2x, l2f,     256,  64,   0.4f, 128, sa3_wi, 3, 256, l3x, l3f);
    sa_stage(l3x, l3f,     64,   16,   0.8f, 256, sa4_wi, 3, 512, l4x, l4f);

    fp_stage(l3x, 64,   l4x, 16,   l3f, 256, l4f, 512, fp4_wi, 2, f3);
    fp_stage(l2x, 256,  l3x, 64,   l2f, 128, f3,  256, fp3_wi, 2, f2_);
    fp_stage(l1x, 1024, l2x, 256,  l1f, 64,  f2_, 256, fp2_wi, 2, f1_);
    fp_stage(xyz, NPTS, l1x, 1024, nullptr, 0, f1_, 128, fp1_wi, 4, f0);

    // head (N padded 13->16, logits stride 16) + log_softmax -> out1
    int rows = Bc * NPTS;
    gemm(f0, rows, 45, inbuf, 0);
    float* out1 = (float*)d_out;
    float* out2 = out1 + (size_t)rows * 13;
    k_log_softmax<<<(rows + 255) / 256, 256, 0, stream>>>(inbuf, out1, 13, 16, rows);
    int t2 = Bc * 512 * 16;
    k_out2<<<(t2 + 255) / 256, 256, 0, stream>>>(l4f, out2, t2);
}
